// ContextualAttention_807453852336
// MI455X (gfx1250) — matrix operand order repaired and checked by reading, not compiled, not run on hardware
//
#include <hip/hip_runtime.h>
#include <math.h>
#include <stdint.h>

// ContextualAttention for MI455X (gfx1250, wave32).
// GEMMs run on v_wmma_f32_16x16x32_bf16 via fp32->bf16 hi/lo split (bf16x3).
// Workspace requirement: 2*B*L + 2*B*L*L floats = ~170 MB in d_ws.

#define BATCH 4
#define CCH   96
#define HH    96
#define WW    96
#define HD    48
#define WD    48
#define LTOT  (HD*WD)        // 2304
#define KSIM  (CCH*9)        // 864  = 27 * 32
#define KDEC  (4*LTOT)       // 9216 = 288 * 32
#define EPSV  1e-9f
#define SCL   10.0f

typedef float  v8f   __attribute__((ext_vector_type(8)));
typedef __bf16 v16bf __attribute__((ext_vector_type(16)));

__device__ __forceinline__ v8f wmma_bf16(v16bf a, v16bf b, v8f c) {
  // D(16x16 f32) = A(16x32 bf16) * B(32x16 bf16) + C ; f32 accumulate.
  return __builtin_amdgcn_wmma_f32_16x16x32_bf16(false, a, false, b, (short)0,
                                                 c, false, false);
}

// Split two f32 into packed bf16 hi words and bf16 lo words (RNE; lo = x-hi).
__device__ __forceinline__ void split2(float a, float b, uint32_t& hi,
                                       uint32_t& lo) {
  __bf16 ah = (__bf16)a, bh = (__bf16)b;
  __bf16 al = (__bf16)(a - (float)ah), bl = (__bf16)(b - (float)bh);
  union { __bf16 h[2]; uint32_t u; } ch, cl;
  ch.h[0] = ah; ch.h[1] = bh;
  cl.h[0] = al; cl.h[1] = bl;
  hi = ch.u; lo = cl.u;
}

// A-fragment (16x32 bf16): VGPR v<4 -> K-pair word v+4*half, v>=4 -> 8+(v-4)+4*half
__device__ __forceinline__ v16bf frag_A(const uint32_t* rowp, int half) {
  v16bf f;
  union { uint32_t u; __bf16 h2[2]; } c;
#pragma unroll
  for (int v = 0; v < 8; ++v) {
    int w = (v < 4) ? (v + 4 * half) : (v + 4 + 4 * half);
    c.u = rowp[w];
    f[2 * v] = c.h2[0]; f[2 * v + 1] = c.h2[1];
  }
  return f;
}

// B-fragment (32x16 bf16): VGPR v<4 -> K-pair word v+4*half, v>=4 -> 8+(v-4)+4*half
__device__ __forceinline__ v16bf frag_B(const uint32_t* rowp, int half) {
  v16bf f;
  union { uint32_t u; __bf16 h2[2]; } c;
#pragma unroll
  for (int v = 0; v < 8; ++v) {
    c.u = rowp[(v < 4) ? (v + 4 * half) : (v + 4 + 4 * half)];
    f[2 * v] = c.h2[0]; f[2 * v + 1] = c.h2[1];
  }
  return f;
}

// 3x3xC patch gather from a stride-2 downsampled plane, zero-padded.
__device__ __forceinline__ float gather_patch(const float* __restrict__ base,
                                              int pos, int k) {
  int c = k / 9, r = k - c * 9;
  int di = r / 3 - 1, dj = (r - (r / 3) * 3) - 1;
  int py = pos / WD, px = pos - py * WD;
  int y = py + di, x = px + dj;
  float v = 0.f;
  if (y >= 0 && y < HD && x >= 0 && x < WD)
    v = base[(size_t)c * HH * WW + (2 * y) * WW + 2 * x];
  return v;
}

// ---------------------------------------------------------------------------
// K1: per-(b,l) inverse patch norm of bgd 3x3xC patch, and mask patch mean.
// norm = sqrt(sum(w^2) + 864*EPS)  (EPS added per element in reference)
// ---------------------------------------------------------------------------
__global__ __launch_bounds__(256) void prep_kernel(
    const float* __restrict__ bg, const float* __restrict__ mask,
    float* __restrict__ inorm, float* __restrict__ mvals) {
  int idx = blockIdx.x * 256 + threadIdx.x;
  if (idx >= BATCH * LTOT) return;
  int b = idx / LTOT, l = idx - b * LTOT;
  int ly = l / WD, lx = l - ly * WD;

  float ss = 0.f;
  for (int c = 0; c < CCH; ++c) {
    const float* pb = bg + (size_t)(b * CCH + c) * HH * WW;
    for (int di = -1; di <= 1; ++di) {
      int y = ly + di; if (y < 0 || y >= HD) continue;
      for (int dj = -1; dj <= 1; ++dj) {
        int x = lx + dj; if (x < 0 || x >= WD) continue;
        float v = pb[(2 * y) * WW + 2 * x];
        ss += v * v;
      }
    }
  }
  inorm[idx] = 1.0f / sqrtf(ss + (float)KSIM * EPSV);

  float ms = 0.f;
  const float* pm = mask + (size_t)b * HH * WW;
  for (int di = -1; di <= 1; ++di) {
    int y = ly + di; if (y < 0 || y >= HD) continue;
    for (int dj = -1; dj <= 1; ++dj) {
      int x = lx + dj; if (x < 0 || x >= WD) continue;
      ms += pm[(2 * y) * WW + 2 * x];
    }
  }
  mvals[idx] = ms * (1.0f / 9.0f);
}

// ---------------------------------------------------------------------------
// K2: sim[l,q] = inorm[l] * sum_k Wpatch[l,k] * Fpatch[q,k], K = 864.
// 64x64 block tile, 8 waves x 2 subtiles, bf16x3 fp32-emulated WMMA.
// ---------------------------------------------------------------------------
__global__ __launch_bounds__(256) void sim_gemm_kernel(
    const float* __restrict__ fg, const float* __restrict__ bg,
    const float* __restrict__ inorm, float* __restrict__ simOut) {
  __shared__ uint32_t WtH[64][17];   // packed 2xbf16 per word, +1 word pad
  __shared__ uint32_t WtL[64][17];
  __shared__ uint32_t FtH[64][17];
  __shared__ uint32_t FtL[64][17];
  __shared__ float sInv[64];

  const int l0 = blockIdx.x * 64;
  const int q0 = blockIdx.y * 64;
  const int b  = blockIdx.z;
  const int tid = threadIdx.x;
  const int lane = tid & 31, wv = tid >> 5;
  const int half = lane >> 4, lr = lane & 15;

  if (tid < 64) sInv[tid] = inorm[b * LTOT + l0 + tid];

  const float* bgB = bg + (size_t)b * CCH * HH * WW;
  const float* fgB = fg + (size_t)b * CCH * HH * WW;

  v8f acc[2] = {{}, {}};

  for (int ks = 0; ks < KSIM; ks += 32) {
#pragma unroll
    for (int p = 0; p < 4; ++p) {        // stage 64 rows x 16 words, W and F
      int e = tid + p * 256;
      int row = e >> 4, kw = e & 15;
      int k = ks + 2 * kw;
      float w0 = gather_patch(bgB, l0 + row, k);
      float w1 = gather_patch(bgB, l0 + row, k + 1);
      float f0 = gather_patch(fgB, q0 + row, k);
      float f1 = gather_patch(fgB, q0 + row, k + 1);
      uint32_t h, lo;
      split2(w0, w1, h, lo); WtH[row][kw] = h; WtL[row][kw] = lo;
      split2(f0, f1, h, lo); FtH[row][kw] = h; FtL[row][kw] = lo;
    }
    __syncthreads();
#pragma unroll
    for (int si = 0; si < 2; ++si) {
      int st = wv + 8 * si;              // 16 subtiles over 8 waves
      int mt = st >> 2, nt = st & 3;
      v16bf ah = frag_A(&WtH[mt * 16 + lr][0], half);
      v16bf al = frag_A(&WtL[mt * 16 + lr][0], half);
      v16bf bh = frag_B(&FtH[nt * 16 + lr][0], half);
      v16bf bl = frag_B(&FtL[nt * 16 + lr][0], half);
      acc[si] = wmma_bf16(ah, bh, acc[si]);   // hi*hi
      acc[si] = wmma_bf16(ah, bl, acc[si]);   // hi*lo
      acc[si] = wmma_bf16(al, bh, acc[si]);   // lo*hi  (lo*lo < 2^-16: dropped)
    }
    __syncthreads();
  }

  float* S = simOut + (size_t)b * LTOT * LTOT;
#pragma unroll
  for (int si = 0; si < 2; ++si) {
    int st = wv + 8 * si;
    int mt = st >> 2, nt = st & 3;
#pragma unroll
    for (int r = 0; r < 8; ++r) {
      int M = r + 8 * half;              // D: vgpr r holds rows r and r+8
      int lrow = l0 + mt * 16 + M;
      int qcol = q0 + nt * 16 + lr;
      S[(size_t)lrow * LTOT + qcol] = acc[si][r] * sInv[mt * 16 + M];
    }
  }
}

// ---------------------------------------------------------------------------
// K3: diagonal fuse in flattened (l,q): out[i,j]=in[i-1,j-1]+in[i,j]+in[i+1,j+1]
// ---------------------------------------------------------------------------
__global__ __launch_bounds__(256) void fuse1_kernel(
    const float* __restrict__ in, float* __restrict__ out) {
  int idx = blockIdx.x * 256 + threadIdx.x;
  int b = idx / (LTOT * LTOT);
  int rem = idx - b * (LTOT * LTOT);
  int i = rem / LTOT, j = rem - i * LTOT;
  const float* p = in + (size_t)b * LTOT * LTOT;
  float v = p[(size_t)i * LTOT + j];
  if (i > 0 && j > 0) v += p[(size_t)(i - 1) * LTOT + (j - 1)];
  if (i < LTOT - 1 && j < LTOT - 1) v += p[(size_t)(i + 1) * LTOT + (j + 1)];
  out[idx] = v;
}

// ---------------------------------------------------------------------------
// K4: same fuse in (x*48+y)-transposed flattened order; I/O in standard order.
// ---------------------------------------------------------------------------
__global__ __launch_bounds__(256) void fuse2_kernel(
    const float* __restrict__ in, float* __restrict__ out) {
  int idx = blockIdx.x * 256 + threadIdx.x;
  int b = idx / (LTOT * LTOT);
  int rem = idx - b * (LTOT * LTOT);
  int l = rem / LTOT, q = rem - l * LTOT;
  int yl = l / WD, xl = l - yl * WD;
  int yq = q / WD, xq = q - yq * WD;
  int ip = xl * WD + yl;                 // transposed-flat row index
  int jp = xq * WD + yq;                 // transposed-flat col index
  const float* p = in + (size_t)b * LTOT * LTOT;
  float v = p[(size_t)l * LTOT + q];
  if (ip > 0 && jp > 0) {
    int im = ip - 1, jm = jp - 1;
    int lm = (im % WD) * WD + (im / WD);
    int qm = (jm % WD) * WD + (jm / WD);
    v += p[(size_t)lm * LTOT + qm];
  }
  if (ip < LTOT - 1 && jp < LTOT - 1) {
    int i1 = ip + 1, j1 = jp + 1;
    int lp = (i1 % WD) * WD + (i1 / WD);
    int qp = (j1 % WD) * WD + (j1 / WD);
    v += p[(size_t)lp * LTOT + qp];
  }
  out[idx] = v;
}

// ---------------------------------------------------------------------------
// K5: in-place column softmax over l: p = softmax(s*m*SCALE, axis=l) * m.
// ---------------------------------------------------------------------------
__global__ __launch_bounds__(256) void softmax_kernel(
    float* __restrict__ S, const float* __restrict__ mvals) {
  __shared__ float sm[LTOT];
  const int bpb = LTOT / 256;            // 9 blocks per sample
  int b = blockIdx.x / bpb;
  int q = (blockIdx.x - b * bpb) * 256 + threadIdx.x;
  for (int i = threadIdx.x; i < LTOT; i += 256) sm[i] = mvals[b * LTOT + i];
  __syncthreads();
  float* P = S + (size_t)b * LTOT * LTOT;
  float mrun = -INFINITY, srun = 0.f;
  for (int l = 0; l < LTOT; ++l) {
    float z = P[(size_t)l * LTOT + q] * sm[l] * SCL;
    if (z > mrun) { srun = srun * __expf(mrun - z) + 1.0f; mrun = z; }
    else          { srun += __expf(z - mrun); }
  }
  float inv = 1.0f / srun;
  for (int l = 0; l < LTOT; ++l) {
    size_t o = (size_t)l * LTOT + q;
    float z = P[o] * sm[l] * SCL;
    P[o] = __expf(z - mrun) * inv * sm[l];
  }
}

// ---------------------------------------------------------------------------
// K6: transposed conv as 4 parity-class GEMMs (M=96, N=2304, K=4*2304):
// y[c,2m+p,2n+q] = 1/16 * sum_l sum_{dy,dx} sim[l,m-dy,n-dx]*wr[l,c,p+2dy,q+2dx]
// wr[l,c,ta,tb] = bg[c, 2*ly+2*ta-2, 2*lx+2*tb-2]  (zero OOB).
// Block: full M=96 x N=64 tile, 8 waves x 3 subtiles, bf16x3 WMMA.
// ---------------------------------------------------------------------------
__global__ __launch_bounds__(256) void deconv_gemm_kernel(
    const float* __restrict__ bg, const float* __restrict__ S,
    float* __restrict__ out) {
  __shared__ uint32_t AtH[CCH][17];
  __shared__ uint32_t AtL[CCH][17];
  __shared__ uint32_t BtH[64][17];
  __shared__ uint32_t BtL[64][17];

  const int s0 = blockIdx.x * 64;
  const int cls = blockIdx.y;
  const int pp = cls >> 1, qq = cls & 1;
  const int b = blockIdx.z;
  const int tid = threadIdx.x;
  const int lane = tid & 31, wv = tid >> 5;
  const int half = lane >> 4, lr = lane & 15;
  const float* simP = S + (size_t)b * LTOT * LTOT;

  v8f acc[3] = {{}, {}, {}};

  for (int ks = 0; ks < KDEC; ks += 32) {
#pragma unroll
    for (int p = 0; p < 6; ++p) {        // stage A: 96 rows x 16 words from bg
      int e = tid + p * 256;
      int row = e >> 4, kw = e & 15;
      int k = ks + 2 * kw;               // pair (k,k+1): same l, taps t,t+1
      int l = k >> 2, t = k & 3;
      int dy = t >> 1;
      int ly = l / WD, lx = l - ly * WD;
      int ry = 2 * ly + 2 * (pp + 2 * dy) - 2;
      int rx0 = 2 * lx + 2 * qq - 2;     // t even -> dx=0
      int rx1 = rx0 + 4;                 // t+1   -> dx=1
      float v0 = 0.f, v1 = 0.f;
      if (ry >= 0 && ry < HH) {
        const float* pr = bg + ((size_t)(b * CCH + row) * HH + ry) * WW;
        if (rx0 >= 0 && rx0 < WW) v0 = pr[rx0];
        if (rx1 >= 0 && rx1 < WW) v1 = pr[rx1];
      }
      uint32_t h, lo;
      split2(v0, v1, h, lo); AtH[row][kw] = h; AtL[row][kw] = lo;
    }
#pragma unroll
    for (int p = 0; p < 4; ++p) {        // stage B: 64 cols x 16 words (shifted sim)
      int e = tid + p * 256;
      int scol = e >> 4, kw = e & 15;
      int k = ks + 2 * kw;
      int l = k >> 2, t = k & 3;
      int dy = t >> 1;
      int s = s0 + scol;
      int mm = s / WD, nn = s - mm * WD;
      int ym = mm - dy;
      float v0 = 0.f, v1 = 0.f;
      if (ym >= 0) {
        const float* pr = simP + (size_t)l * LTOT + ym * WD;
        v0 = pr[nn];                              // dx = 0
        if (nn >= 1) v1 = pr[nn - 1];             // dx = 1
      }
      uint32_t h, lo;
      split2(v0, v1, h, lo); BtH[scol][kw] = h; BtL[scol][kw] = lo;
    }
    __syncthreads();
#pragma unroll
    for (int si = 0; si < 3; ++si) {
      int st = wv + 8 * si;              // 24 subtiles over 8 waves
      int mt = st >> 2, nt = st & 3;
      v16bf ah = frag_A(&AtH[mt * 16 + lr][0], half);
      v16bf al = frag_A(&AtL[mt * 16 + lr][0], half);
      v16bf bh = frag_B(&BtH[nt * 16 + lr][0], half);
      v16bf bl = frag_B(&BtL[nt * 16 + lr][0], half);
      acc[si] = wmma_bf16(ah, bh, acc[si]);
      acc[si] = wmma_bf16(ah, bl, acc[si]);
      acc[si] = wmma_bf16(al, bh, acc[si]);
    }
    __syncthreads();
  }

  const float r16 = 1.0f / 16.0f;
#pragma unroll
  for (int si = 0; si < 3; ++si) {
    int st = wv + 8 * si;
    int mt = st >> 2, nt = st & 3;
#pragma unroll
    for (int r = 0; r < 8; ++r) {
      int M = r + 8 * half;
      int c = mt * 16 + M;
      int s = s0 + nt * 16 + lr;
      int mm = s / WD, nn = s - mm * WD;
      int oh = 2 * mm + pp, ow = 2 * nn + qq;
      out[((b * CCH + c) * HH + oh) * WW + ow] = acc[si][r] * r16;
    }
  }
}

// ---------------------------------------------------------------------------
extern "C" void kernel_launch(void* const* d_in, const int* in_sizes, int n_in,
                              void* d_out, int out_size, void* d_ws, size_t ws_size,
                              hipStream_t stream) {
  (void)in_sizes; (void)n_in; (void)out_size; (void)ws_size;
  const float* fg   = (const float*)d_in[0];
  const float* bg   = (const float*)d_in[1];
  const float* mask = (const float*)d_in[2];
  float* out = (float*)d_out;

  float* ws    = (float*)d_ws;
  float* inorm = ws;                                   // B*L
  float* mvals = ws + BATCH * LTOT;                    // B*L
  float* bufA  = ws + 2 * BATCH * LTOT;                // B*L*L
  float* bufB  = bufA + (size_t)BATCH * LTOT * LTOT;   // B*L*L

  prep_kernel<<<dim3((BATCH * LTOT + 255) / 256), 256, 0, stream>>>(
      bg, mask, inorm, mvals);
  sim_gemm_kernel<<<dim3(LTOT / 64, LTOT / 64, BATCH), 256, 0, stream>>>(
      fg, bg, inorm, bufA);
  const int nel = BATCH * LTOT * LTOT;
  fuse1_kernel<<<dim3(nel / 256), 256, 0, stream>>>(bufA, bufB);
  fuse2_kernel<<<dim3(nel / 256), 256, 0, stream>>>(bufB, bufA);
  softmax_kernel<<<dim3(BATCH * (LTOT / 256)), 256, 0, stream>>>(bufA, mvals);
  deconv_gemm_kernel<<<dim3(LTOT / 64, 4, BATCH), 256, 0, stream>>>(
      bg, bufA, out);
}
